// NodeAttention_12317966205315
// MI455X (gfx1250) — matrix operand order repaired and checked by reading, not compiled, not run on hardware
//
#include <hip/hip_runtime.h>
#include <hip/hip_bf16.h>

// ---------------------------------------------------------------------------
// Types for CDNA5 WMMA (wave32, 16x16x32 bf16 -> f32)
// ---------------------------------------------------------------------------
typedef __attribute__((ext_vector_type(16))) __bf16 bf16x16;
typedef __attribute__((ext_vector_type(8)))  float  f32x8;

#define H 256
#define NUM_SEGMENTS 1024

// Hardware tanh on gfx1250 (V_TANH_F32, TRANS op). Try both plausible builtin
// names; otherwise a branch-free v_exp_f32 + v_rcp_f32 fallback.
#if defined(__has_builtin)
#  if __has_builtin(__builtin_amdgcn_tanh_f32)
#    define FAST_TANH(x) __builtin_amdgcn_tanh_f32(x)
#  elif __has_builtin(__builtin_amdgcn_tanhf)
#    define FAST_TANH(x) __builtin_amdgcn_tanhf(x)
#  endif
#endif
#ifndef FAST_TANH
__device__ __forceinline__ float fast_tanh_impl(float v) {
    const float xc = fminf(fmaxf(v, -9.f), 9.f);
    const float t  = __expf(2.f * xc);                       // v_exp_f32
    return (t - 1.f) * __builtin_amdgcn_rcpf(t + 1.f);       // v_rcp_f32
}
#define FAST_TANH(x) fast_tanh_impl(x)
#endif

__device__ __forceinline__ unsigned short f32_to_bf16_bits(float f) {
    unsigned int u = __float_as_uint(f);
    unsigned int r = u + 0x7FFFu + ((u >> 16) & 1u);   // round-to-nearest-even
    return (unsigned short)(r >> 16);
}

// XOR-swizzle of 8-element (16B) chunks by row: spreads the 16 A-fragment rows
// across all 64 LDS banks with a 512B (bank-aliasing) row stride.
#define SWIZ(k, r) ((((((k) >> 3) ^ ((r) & 15)) << 3)) | ((k) & 7))

// ---------------------------------------------------------------------------
// Kernel 1: W1 (fp32, [k][n]) -> W1T (bf16, [n][k]) in workspace.
// ---------------------------------------------------------------------------
__global__ __launch_bounds__(256)
void w1_transpose_bf16(const float* __restrict__ W1, unsigned short* __restrict__ w1t) {
    const int k = blockIdx.x;    // 0..255
    const int n = threadIdx.x;   // 0..255 (coalesced read of W1 row k)
    w1t[(size_t)n * H + k] = f32_to_bf16_bits(W1[(size_t)k * H + n]);
}

// ---------------------------------------------------------------------------
// Kernel 2: scores[N] = tanh(x @ W1 + b1) @ w2 via v_wmma_f32_16x16x32_bf16.
// 128 threads = 4 waves; 128 rows/block; each wave owns 32 rows (2 M-tiles),
// so every B fragment feeds 2 WMMAs. x tile bf16-converted into 64KB LDS,
// A fragments live in registers; B is double-buffered in registers.
// ---------------------------------------------------------------------------
__global__ __launch_bounds__(128)
void score_gemm(const float* __restrict__ x,
                const unsigned short* __restrict__ w1t,
                const float* __restrict__ b1,
                const float* __restrict__ w2,
                float* __restrict__ scores, int N) {
    __shared__ unsigned short xlds[128][256];   // exactly 64 KB, XOR-swizzled

    const int tid = threadIdx.x;
    const int r0  = blockIdx.x * 128;

    // ---- stage: 128 rows x 256 fp32 -> bf16 LDS (float4 granules) ----
    for (int idx = tid; idx < 128 * 64; idx += 128) {
        const int row = idx >> 6;            // 0..127
        const int c4  = (idx & 63) << 2;     // 0,4,...,252 (stays inside a chunk)
        const int gr  = r0 + row;
        float4 v = make_float4(0.f, 0.f, 0.f, 0.f);
        if (gr < N) v = *(const float4*)(x + (size_t)gr * H + c4);
        const int pc = SWIZ(c4, row);
        xlds[row][pc + 0] = f32_to_bf16_bits(v.x);
        xlds[row][pc + 1] = f32_to_bf16_bits(v.y);
        xlds[row][pc + 2] = f32_to_bf16_bits(v.z);
        xlds[row][pc + 3] = f32_to_bf16_bits(v.w);
    }
    __syncthreads();

    const int wv   = tid >> 5;       // wave 0..3 -> rows [wv*32, wv*32+32)
    const int lane = tid & 31;
    const int half = lane >> 4;      // 0 | 1
    const int ln   = lane & 15;      // 0..15

    // A fragment rows for this lane (ISA: A lane stripes M = lane%16).
    // Both rows have (row & 15) == ln, so swizzle index is ln for both.
    const unsigned short* arow0 = &xlds[wv * 32 + ln][0];
    const unsigned short* arow1 = &xlds[wv * 32 + 16 + ln][0];

    // Preload all A fragments (2 M-tiles x 8 k-steps) into registers.
    union AF { bf16x16 v; uint4 q[2]; };
    AF a0[8], a1[8];
    #pragma unroll
    for (int ks = 0; ks < 8; ++ks) {
        const int k0 = ks * 32;
        a0[ks].q[0] = *(const uint4*)(arow0 + SWIZ(k0 + half * 8, ln));
        a0[ks].q[1] = *(const uint4*)(arow0 + SWIZ(k0 + 16 + half * 8, ln));
        a1[ks].q[0] = *(const uint4*)(arow1 + SWIZ(k0 + half * 8, ln));
        a1[ks].q[1] = *(const uint4*)(arow1 + SWIZ(k0 + 16 + half * 8, ln));
    }

    float acc0[8], acc1[8];
    #pragma unroll
    for (int v = 0; v < 8; ++v) { acc0[v] = 0.f; acc1[v] = 0.f; }

    #pragma unroll 1
    for (int n0 = 0; n0 < H; n0 += 16) {
        union CF { f32x8 v; float f[8]; } c0, c1;
        c0.v = (f32x8){0.f,0.f,0.f,0.f,0.f,0.f,0.f,0.f};
        c1.v = (f32x8){0.f,0.f,0.f,0.f,0.f,0.f,0.f,0.f};

        // B column for this lane: n = n0 + ln, K chunks {8h..8h+7, 16+8h..16+8h+7}
        const unsigned short* bcol = w1t + (size_t)(n0 + ln) * H + half * 8;

        // Software-pipelined B: next fragment's loads in flight during WMMAs.
        union BF { bf16x16 v; uint4 q[2]; };
        BF bcur, bnxt;
        bcur.q[0] = *(const uint4*)(bcol + 0);
        bcur.q[1] = *(const uint4*)(bcol + 16);

        #pragma unroll
        for (int ks = 0; ks < 8; ++ks) {
            if (ks < 7) {
                const int kn = (ks + 1) * 32;
                bnxt.q[0] = *(const uint4*)(bcol + kn);
                bnxt.q[1] = *(const uint4*)(bcol + kn + 16);
            }
            c0.v = __builtin_amdgcn_wmma_f32_16x16x32_bf16(
                false, a0[ks].v, false, bcur.v, (short)0, c0.v, false, false);
            c1.v = __builtin_amdgcn_wmma_f32_16x16x32_bf16(
                false, a1[ks].v, false, bcur.v, (short)0, c1.v, false, false);
            if (ks < 7) bcur = bnxt;
        }

        // C layout: lane l holds N = l%16, VGPR v holds M = v + 8*(l/16).
        const float bn = b1[n0 + ln];
        const float wn = w2[n0 + ln];
        #pragma unroll
        for (int v = 0; v < 8; ++v) {
            acc0[v] += FAST_TANH(c0.f[v] + bn) * wn;
            acc1[v] += FAST_TANH(c1.f[v] + bn) * wn;
        }
    }

    // Reduce partial scores across the 16 lanes of each half-wave.
    #pragma unroll
    for (int m = 1; m < 16; m <<= 1) {
        #pragma unroll
        for (int v = 0; v < 8; ++v) {
            acc0[v] += __shfl_xor(acc0[v], m, 32);
            acc1[v] += __shfl_xor(acc1[v], m, 32);
        }
    }

    if (ln == 0) {
        const int base0 = r0 + wv * 32 + half * 8;        // tile 0: rows +0..15
        const int base1 = r0 + wv * 32 + 16 + half * 8;   // tile 1: rows +16..31
        #pragma unroll
        for (int v = 0; v < 8; ++v) {
            if (base0 + v < N) scores[base0 + v] = acc0[v];
            if (base1 + v < N) scores[base1 + v] = acc1[v];
        }
    }
}

// ---------------------------------------------------------------------------
// Kernel 3: per-segment max and sum(exp(score - max)). batch sorted int64.
// ---------------------------------------------------------------------------
__device__ __forceinline__ int lower_bound_ll(const long long* __restrict__ a,
                                              int n, long long v) {
    int lo = 0, hi = n;
    while (lo < hi) {
        int mid = (lo + hi) >> 1;
        if (a[mid] < v) lo = mid + 1; else hi = mid;
    }
    return lo;
}

__global__ __launch_bounds__(256)
void segment_stats(const float* __restrict__ scores,
                   const long long* __restrict__ batch, int N,
                   float* __restrict__ segmax, float* __restrict__ segsum) {
    __shared__ float red[256];
    __shared__ int bounds[2];
    const int s   = blockIdx.x;
    const int tid = threadIdx.x;

    if (tid == 0) {
        bounds[0] = lower_bound_ll(batch, N, (long long)s);
        bounds[1] = lower_bound_ll(batch, N, (long long)s + 1);
    }
    __syncthreads();
    const int start = bounds[0], end = bounds[1];

    float m = -INFINITY;
    for (int i = start + tid; i < end; i += 256) m = fmaxf(m, scores[i]);
    red[tid] = m;
    __syncthreads();
    for (int st = 128; st > 0; st >>= 1) {
        if (tid < st) red[tid] = fmaxf(red[tid], red[tid + st]);
        __syncthreads();
    }
    const float mx = red[0];
    __syncthreads();

    float sum = 0.f;
    for (int i = start + tid; i < end; i += 256) sum += __expf(scores[i] - mx);
    red[tid] = sum;
    __syncthreads();
    for (int st = 128; st > 0; st >>= 1) {
        if (tid < st) red[tid] += red[tid + st];
        __syncthreads();
    }
    if (tid == 0) { segmax[s] = mx; segsum[s] = red[0]; }
}

// ---------------------------------------------------------------------------
// Kernel 4: zero the 256-float output (d_out is poisoned, must be rewritten).
// ---------------------------------------------------------------------------
__global__ void zero_out(float* __restrict__ out) { out[threadIdx.x] = 0.f; }

// ---------------------------------------------------------------------------
// Kernel 5: out[h] = sum_n x[n][h] * softmax_weight(n). 256 rows per block,
// weights staged in LDS, coalesced column reads, one atomicAdd per column.
// ---------------------------------------------------------------------------
__global__ __launch_bounds__(256)
void weighted_sum(const float* __restrict__ x,
                  const float* __restrict__ scores,
                  const long long* __restrict__ batch,
                  const float* __restrict__ segmax,
                  const float* __restrict__ segsum,
                  float* __restrict__ out, int N) {
    __shared__ float wbuf[256];
    const int t  = threadIdx.x;
    const int r0 = blockIdx.x * 256;
    const int r  = r0 + t;

    float w = 0.f;
    if (r < N) {
        const int s = (int)batch[r];
        w = __expf(scores[r] - segmax[s]) / segsum[s];
    }
    wbuf[t] = w;
    __syncthreads();

    const int rows = (N - r0) < 256 ? (N - r0) : 256;
    float acc = 0.f;
    for (int i = 0; i < rows; ++i) {
        if (i + 8 < rows)
            __builtin_prefetch(x + (size_t)(r0 + i + 8) * H + t, 0, 0);  // global_prefetch_b8
        acc += x[(size_t)(r0 + i) * H + t] * wbuf[i];
    }
    atomicAdd(&out[t], acc);
}

// ---------------------------------------------------------------------------
// Launch
// ---------------------------------------------------------------------------
extern "C" void kernel_launch(void* const* d_in, const int* in_sizes, int n_in,
                              void* d_out, int out_size, void* d_ws, size_t ws_size,
                              hipStream_t stream) {
    const float*     x     = (const float*)d_in[0];     // [N,256] fp32
    const float*     W1    = (const float*)d_in[1];     // [256,256] fp32
    const float*     b1    = (const float*)d_in[2];     // [256] fp32
    const float*     w2    = (const float*)d_in[3];     // [256,1] fp32
    const long long* batch = (const long long*)d_in[4]; // [N] int64, sorted
    const int N = in_sizes[4];

    // Workspace layout: W1T(bf16) | scores | segmax | segsum  (~2.14 MB)
    unsigned short* w1t    = (unsigned short*)d_ws;
    float*          scores = (float*)((char*)d_ws + (size_t)H * H * 2);
    float*          smax   = scores + N;
    float*          ssum   = smax + NUM_SEGMENTS;
    float*          out    = (float*)d_out;

    w1_transpose_bf16<<<H, 256, 0, stream>>>(W1, w1t);
    score_gemm<<<(N + 127) / 128, 128, 0, stream>>>(x, w1t, b1, w2, scores, N);
    segment_stats<<<NUM_SEGMENTS, 256, 0, stream>>>(scores, batch, N, smax, ssum);
    zero_out<<<1, 256, 0, stream>>>(out);
    weighted_sum<<<(N + 255) / 256, 256, 0, stream>>>(x, scores, batch, smax, ssum, out, N);
}
